// GAU_77644418777751
// MI455X (gfx1250) — matrix operand order repaired and checked by reading, not compiled, not run on hardware
//
#include <hip/hip_runtime.h>
#include <hip/hip_bf16.h>
#include <math.h>

typedef __attribute__((ext_vector_type(16))) _Float16 v16h;
typedef __attribute__((ext_vector_type(8)))  _Float16 v8h;
typedef __attribute__((ext_vector_type(4)))  _Float16 v4h;
typedef __attribute__((ext_vector_type(8)))  float    v8f;
typedef int vi4 __attribute__((vector_size(4 * sizeof(int))));

#define BATCH 16
#define CIN   256
#define SDIM  128
#define ODIM  256
#define NPIX  4096      // 64*64 per batch
#define PIXT  64        // pixel tile per block
#define CSTR  264       // LDS row stride in halves (256 + 8 pad)
#define KTILE 32
#define NKT   8         // 256 / 32

// workspace layout (bytes)
#define WS_MU    ((size_t)0)         // 4096 f32
#define WS_RSTD  ((size_t)16384)     // 4096 f32
#define WS_SIM   ((size_t)32768)     // 65536 f32
#define WS_WG    ((size_t)294912)    // 65536 f16
#define WS_WV    ((size_t)425984)    // 65536 f16
#define WS_WZ    ((size_t)557056)    // 32768 f16
#define WS_WP    ((size_t)622592)    // 65536 f16
#define WS_P     ((size_t)753664)    // 16*256*4096 f16 (g*v product)

// dynamic LDS sizes (bytes)
#define LDS_MAIN ((PIXT * CSTR + 2 * 32 * CSTR) * 2)   // xn tile + 2x(gate|value) bufs
#define LDS_PROJ ((PIXT * CSTR + 2 * 16 * CSTR) * 2)   // V tile  + 2x proj bufs

// ------------------------------------------------ CDNA5 async global->LDS copy
#if defined(__has_builtin)
#if __has_builtin(__builtin_amdgcn_global_load_async_to_lds_b128)
#define HAVE_ASYNC 1
#endif
#endif
#ifndef HAVE_ASYNC
#define HAVE_ASYNC 0
#endif

#if HAVE_ASYNC
typedef __attribute__((address_space(1))) vi4* gptr4;
typedef __attribute__((address_space(3))) vi4* lptr4;
#endif

__device__ __forceinline__ void cp16_async(_Float16* l, const _Float16* g) {
#if HAVE_ASYNC
  __builtin_amdgcn_global_load_async_to_lds_b128(
      (gptr4)(void*)g, (lptr4)(void*)l, 0, 0);
#else
  *(v8h*)l = *(const v8h*)g;
#endif
}

__device__ __forceinline__ void wait_async() {
#if HAVE_ASYNC
#if __has_builtin(__builtin_amdgcn_s_wait_asynccnt)
  __builtin_amdgcn_s_wait_asynccnt(0);
#else
  asm volatile("s_wait_asynccnt 0" ::: "memory");
#endif
#endif
}

// Stage a 16x256 f16 weight tile (rows [mrow,mrow+16)) into LDS, stride CSTR.
// 512 16-byte chunks over 128 threads.
__device__ __forceinline__ void stage_wtile(_Float16* dst,
                                            const _Float16* __restrict__ W,
                                            int mrow, int tid) {
#pragma unroll
  for (int j = 0; j < 4; ++j) {
    int id  = j * 128 + tid;
    int row = id >> 5;         // 32 chunks per row
    int c8  = (id & 31) * 8;   // half offset within row
    cp16_async(dst + row * CSTR + c8, W + (size_t)(mrow + row) * CIN + c8);
  }
}

// ---------------------------------------------------------------- WMMA helpers
__device__ __forceinline__ v8f wmma32(v16h a, v16h b, v8f c) {
  return __builtin_amdgcn_wmma_f32_16x16x32_f16(
      false, a, false, b, (short)0, c, false, false);
}

// A fragment 16x32 f16 from an LDS weight tile (stride CSTR).
__device__ __forceinline__ v16h lds_afrag(const _Float16* __restrict__ wt,
                                          int k0, int lane) {
  int m  = lane & 15;
  int kb = (lane & 16) ? 16 : 0;
  const _Float16* p = wt + m * CSTR + k0 + kb;
  v8h lo = *(const v8h*)p;          // K = kb .. kb+7
  v8h hi = *(const v8h*)(p + 8);    // K = kb+8 .. kb+15
  v16h a;
#pragma unroll
  for (int j = 0; j < 8; ++j) { a[j] = lo[j]; a[j + 8] = hi[j]; }
  return a;
}

// B fragment 32x16 f16 from transposed x_n LDS row (channels contiguous).
__device__ __forceinline__ v16h load_bfrag(const _Float16* __restrict__ row,
                                           int k0, int lane) {
  int kof = (lane & 16) ? 16 : 0;
  return *(const v16h*)(row + k0 + kof);
}

// ---------------------------------------------------------------- kernel 1: stats
__global__ void __launch_bounds__(256)
gau_stats(const float* __restrict__ x, float* __restrict__ mu,
          float* __restrict__ rstd) {
  __shared__ float s1[256], s2[256];
  int bc = blockIdx.x;
  const float4* p = (const float4*)(x + (size_t)bc * NPIX);
  float sum = 0.f, sq = 0.f;
#pragma unroll
  for (int i = 0; i < 4; ++i) {
    float4 v = p[threadIdx.x + i * 256];
    sum += v.x + v.y + v.z + v.w;
    sq  += v.x * v.x + v.y * v.y + v.z * v.z + v.w * v.w;
  }
  s1[threadIdx.x] = sum; s2[threadIdx.x] = sq;
  __syncthreads();
  for (int off = 128; off > 0; off >>= 1) {
    if (threadIdx.x < off) {
      s1[threadIdx.x] += s1[threadIdx.x + off];
      s2[threadIdx.x] += s2[threadIdx.x + off];
    }
    __syncthreads();
  }
  if (threadIdx.x == 0) {
    float m   = s1[0] * (1.f / NPIX);
    float var = s2[0] * (1.f / NPIX) - m * m;
    mu[bc]   = m;
    rstd[bc] = rsqrtf(var + 1e-5f);
  }
}

// ---------------------------------------------------------------- kernel 2: weights -> f16
__global__ void __launch_bounds__(256)
gau_wconv(const float* __restrict__ gW, const float* __restrict__ vW,
          const float* __restrict__ zW, const float* __restrict__ pW,
          _Float16* __restrict__ wg, _Float16* __restrict__ wv,
          _Float16* __restrict__ wz, _Float16* __restrict__ wp) {
  int i = blockIdx.x * 256 + threadIdx.x;
  if (i < CIN * CIN)  wg[i] = (_Float16)gW[i];
  if (i < CIN * CIN)  wv[i] = (_Float16)vW[i];
  if (i < SDIM * CIN) wz[i] = (_Float16)zW[i];
  if (i < ODIM * CIN) wp[i] = (_Float16)pW[i];
}

// ---------------------------------------------------------------- kernel 3: fused main
__global__ void __launch_bounds__(128)
gau_main(const float* __restrict__ x,
         const float* __restrict__ gate_b, const float* __restrict__ value_b,
         const float* __restrict__ z_b,
         const float* __restrict__ gamma, const float* __restrict__ beta,
         const float* __restrict__ mu, const float* __restrict__ rstd,
         const _Float16* __restrict__ Wg, const _Float16* __restrict__ Wv,
         const _Float16* __restrict__ Wz,
         _Float16* __restrict__ P, float* __restrict__ sim) {
  extern __shared__ _Float16 smem[];
  _Float16* ldsT = smem;                      // [PIXT][CSTR] x_n transposed
  _Float16* wb0  = smem + PIXT * CSTR;        // 32 rows (gate|value) buf 0
  _Float16* wb1  = wb0 + 32 * CSTR;           // buf 1

  int b    = blockIdx.x >> 6;
  int tile = blockIdx.x & 63;
  int pix0 = tile * PIXT;
  int tid  = threadIdx.x;

  const float* xb = x + (size_t)b * CIN * NPIX;
#pragma unroll 4
  for (int it = 0; it < 32; ++it) {           // normalize 256ch x 64pix -> f16
    int lin = it * 128 + tid;
    int c   = lin >> 4;
    int p4  = (lin & 15) * 4;
    float m = mu[b * CIN + c], r = rstd[b * CIN + c];
    float4 v = *(const float4*)(xb + (size_t)c * NPIX + pix0 + p4);
    ldsT[(p4 + 0) * CSTR + c] = (_Float16)((v.x - m) * r);
    ldsT[(p4 + 1) * CSTR + c] = (_Float16)((v.y - m) * r);
    ldsT[(p4 + 2) * CSTR + c] = (_Float16)((v.z - m) * r);
    ldsT[(p4 + 3) * CSTR + c] = (_Float16)((v.w - m) * r);
  }
  // prefetch first gate+value weight tile while finishing staging
  stage_wtile(wb0, Wg, 0, tid);
  stage_wtile(wb0 + 16 * CSTR, Wv, 0, tid);
  __syncthreads();

  int wave = tid >> 5, lane = tid & 31;
  int n = lane & 15, hi = (lane >> 4) & 1;
  int nBase = wave * 16;
  int pix = pix0 + nBase + n;
  const _Float16* brow = ldsT + (nBase + n) * CSTR;

  v16h bf[NKT];
#pragma unroll
  for (int k = 0; k < NKT; ++k) bf[k] = load_bfrag(brow, k * KTILE, lane);

  // ---- gate & value tiles -> P = g*v -------------------------------------
  for (int t = 0; t < 16; ++t) {
    wait_async();
    __syncthreads();
    _Float16* cur = (t & 1) ? wb1 : wb0;
    if (t < 15) {
      _Float16* nxt = (t & 1) ? wb0 : wb1;
      stage_wtile(nxt, Wg, (t + 1) * 16, tid);
      stage_wtile(nxt + 16 * CSTR, Wv, (t + 1) * 16, tid);
    }
    int ob = t * 16;
    v8f ag = {}, av = {};
#pragma unroll
    for (int k = 0; k < NKT; ++k) {
      ag = wmma32(lds_afrag(cur, k * KTILE, lane), bf[k], ag);
      av = wmma32(lds_afrag(cur + 16 * CSTR, k * KTILE, lane), bf[k], av);
    }
#pragma unroll
    for (int r = 0; r < 8; ++r) {
      int o = ob + r + 8 * hi;
      float g = ag[r] + gate_b[o];
      float v = av[r] + value_b[o];
      P[(size_t)(b * CIN + o) * NPIX + pix] = (_Float16)(g * v);
    }
  }

  // ---- z tiles -> affine -> RoPE -> q.k ----------------------------------
  stage_wtile(wb0, Wz, 0, tid);
  float simp = 0.f;
  int h = pix >> 6, w = pix & 63;
  for (int t = 0; t < 8; ++t) {
    wait_async();
    __syncthreads();
    _Float16* cur = (t & 1) ? wb1 : wb0;
    if (t < 7) stage_wtile((t & 1) ? wb0 : wb1, Wz, (t + 1) * 16, tid);
    int sb = t * 16;
    v8f az = {};
#pragma unroll
    for (int k = 0; k < NKT; ++k)
      az = wmma32(lds_afrag(cur, k * KTILE, lane), bf[k], az);
#pragma unroll
    for (int r = 0; r < 8; r += 2) {          // channel pairs (s0, s0+1) in-lane
      int s0 = sb + r + 8 * hi;
      float z1 = az[r]     + z_b[s0];
      float z2 = az[r + 1] + z_b[s0 + 1];
      float q1 = z1 * gamma[s0]     + beta[s0];
      float q2 = z2 * gamma[s0 + 1] + beta[s0 + 1];
      float k1 = z1 * gamma[SDIM + s0]     + beta[SDIM + s0];
      float k2 = z2 * gamma[SDIM + s0 + 1] + beta[SDIM + s0 + 1];
      int  i   = s0 >> 1;
      float pos = (i < 32) ? (float)h : (float)w;
      float f   = (float)((i < 32) ? i : i - 32);
      float ang = pos * exp2f(f * -0.4152410118609203f);  // 10000^(-f/32)
      float sn, cs;
      __sincosf(ang, &sn, &cs);
      float q1r = q1 * cs - q2 * sn, q2r = q1 * sn + q2 * cs;
      float k1r = k1 * cs - k2 * sn, k2r = k1 * sn + k2 * cs;
      simp += q1r * k1r + q2r * k2r;
    }
  }
  simp += __shfl_xor(simp, 16, 32);
  if (lane < 16)
    sim[(size_t)b * NPIX + pix] = simp * 0.08838834764831845f;  // 1/sqrt(128)
}

// ---------------------------------------------------------------- kernel 4: softmax
__global__ void __launch_bounds__(256)
gau_softmax(const float* __restrict__ sim, float* __restrict__ A) {
  __shared__ float red[256];
  int b = blockIdx.x, tid = threadIdx.x;
  const float* s = sim + (size_t)b * NPIX;
  float v[16];
  float m = -1e30f;
#pragma unroll
  for (int i = 0; i < 16; ++i) { v[i] = s[tid + i * 256]; m = fmaxf(m, v[i]); }
  red[tid] = m; __syncthreads();
  for (int off = 128; off > 0; off >>= 1) {
    if (tid < off) red[tid] = fmaxf(red[tid], red[tid + off]);
    __syncthreads();
  }
  m = red[0]; __syncthreads();
  float sum = 0.f;
#pragma unroll
  for (int i = 0; i < 16; ++i) { v[i] = __expf(v[i] - m); sum += v[i]; }
  red[tid] = sum; __syncthreads();
  for (int off = 128; off > 0; off >>= 1) {
    if (tid < off) red[tid] += red[tid + off];
    __syncthreads();
  }
  float inv = 1.f / red[0];
#pragma unroll
  for (int i = 0; i < 16; ++i)
    A[(size_t)b * NPIX + tid + i * 256] = v[i] * inv;
}

// ---------------------------------------------------------------- kernel 5: epilogue
__global__ void __launch_bounds__(128)
gau_proj(const float* __restrict__ x, const float* __restrict__ mu,
         const float* __restrict__ rstd, const _Float16* __restrict__ P,
         const float* __restrict__ A, const _Float16* __restrict__ Wp,
         const float* __restrict__ proj_b, float* __restrict__ out) {
  extern __shared__ _Float16 smem[];
  _Float16* ldsT = smem;
  _Float16* wb0  = smem + PIXT * CSTR;
  _Float16* wb1  = wb0 + 16 * CSTR;

  int b    = blockIdx.x >> 6;
  int tile = blockIdx.x & 63;
  int pix0 = tile * PIXT;
  int tid  = threadIdx.x;

  const float*    xb = x + (size_t)b * CIN * NPIX;
  const _Float16* Pb = P + (size_t)b * CIN * NPIX;
  const float*    Ab = A + (size_t)b * NPIX;
#pragma unroll 4
  for (int it = 0; it < 32; ++it) {           // V = A*P + x_n  (f16)
    int lin = it * 128 + tid;
    int c   = lin >> 4;
    int p4  = (lin & 15) * 4;
    float m = mu[b * CIN + c], r = rstd[b * CIN + c];
    float4 v = *(const float4*)(xb + (size_t)c * NPIX + pix0 + p4);
    v4h pv = *(const v4h*)(Pb + (size_t)c * NPIX + pix0 + p4);
    ldsT[(p4 + 0) * CSTR + c] = (_Float16)(Ab[pix0 + p4 + 0] * (float)pv[0] + (v.x - m) * r);
    ldsT[(p4 + 1) * CSTR + c] = (_Float16)(Ab[pix0 + p4 + 1] * (float)pv[1] + (v.y - m) * r);
    ldsT[(p4 + 2) * CSTR + c] = (_Float16)(Ab[pix0 + p4 + 2] * (float)pv[2] + (v.z - m) * r);
    ldsT[(p4 + 3) * CSTR + c] = (_Float16)(Ab[pix0 + p4 + 3] * (float)pv[3] + (v.w - m) * r);
  }
  stage_wtile(wb0, Wp, 0, tid);
  __syncthreads();

  int wave = tid >> 5, lane = tid & 31;
  int n = lane & 15, hi = (lane >> 4) & 1;
  int nBase = wave * 16;
  int pix = pix0 + nBase + n;
  const _Float16* brow = ldsT + (nBase + n) * CSTR;

  v16h bf[NKT];
#pragma unroll
  for (int k = 0; k < NKT; ++k) bf[k] = load_bfrag(brow, k * KTILE, lane);

  for (int t = 0; t < 16; ++t) {
    wait_async();
    __syncthreads();
    _Float16* cur = (t & 1) ? wb1 : wb0;
    if (t < 15) stage_wtile((t & 1) ? wb0 : wb1, Wp, (t + 1) * 16, tid);
    int ob = t * 16;
    v8f acc = {};
#pragma unroll
    for (int k = 0; k < NKT; ++k)
      acc = wmma32(lds_afrag(cur, k * KTILE, lane), bf[k], acc);
#pragma unroll
    for (int r = 0; r < 8; ++r) {
      int o = ob + r + 8 * hi;
      out[(size_t)(b * ODIM + o) * NPIX + pix] = acc[r] + proj_b[o];
    }
  }
}

// ---------------------------------------------------------------- launch
extern "C" void kernel_launch(void* const* d_in, const int* in_sizes, int n_in,
                              void* d_out, int out_size, void* d_ws, size_t ws_size,
                              hipStream_t stream) {
  const float* x       = (const float*)d_in[0];
  const float* gate_W  = (const float*)d_in[1];
  const float* gate_b  = (const float*)d_in[2];
  const float* value_W = (const float*)d_in[3];
  const float* value_b = (const float*)d_in[4];
  const float* z_W     = (const float*)d_in[5];
  const float* z_b     = (const float*)d_in[6];
  const float* gamma   = (const float*)d_in[7];
  const float* beta    = (const float*)d_in[8];
  const float* proj_W  = (const float*)d_in[9];
  const float* proj_b  = (const float*)d_in[10];

  char* ws = (char*)d_ws;
  float*    mu   = (float*)(ws + WS_MU);
  float*    rstd = (float*)(ws + WS_RSTD);
  float*    sim  = (float*)(ws + WS_SIM);
  _Float16* wg   = (_Float16*)(ws + WS_WG);
  _Float16* wv   = (_Float16*)(ws + WS_WV);
  _Float16* wz   = (_Float16*)(ws + WS_WZ);
  _Float16* wp   = (_Float16*)(ws + WS_WP);
  _Float16* Pws  = (_Float16*)(ws + WS_P);

  float* outMain = (float*)d_out;                              // [16,256,64,64]
  float* Aout    = outMain + (size_t)BATCH * ODIM * NPIX;      // [16,64,64]

  gau_stats<<<BATCH * CIN, 256, 0, stream>>>(x, mu, rstd);
  gau_wconv<<<256, 256, 0, stream>>>(gate_W, value_W, z_W, proj_W, wg, wv, wz, wp);
  gau_main<<<BATCH * (NPIX / PIXT), 128, LDS_MAIN, stream>>>(
      x, gate_b, value_b, z_b, gamma, beta, mu, rstd, wg, wv, wz, Pws, sim);
  gau_softmax<<<BATCH, 256, 0, stream>>>(sim, Aout);
  gau_proj<<<BATCH * (NPIX / PIXT), 128, LDS_PROJ, stream>>>(
      x, mu, rstd, Pws, Aout, wp, proj_b, outMain);
}
